// Packer_88029649699049
// MI455X (gfx1250) — compile-verified, matrix operand order checked by reading, not run on hardware
//
#include <hip/hip_runtime.h>
#include <hip/hip_bf16.h>
#include <math.h>

#define BB 2
#define LL 1024
#define KK 30
#define MM 16
#define NRBF 16

typedef __attribute__((ext_vector_type(16))) _Float16 v16h;
typedef __attribute__((ext_vector_type(8)))  float    v8f;

// ---------------- output layout (floats, concatenated in return order) ----
static constexpr long OFF_V    = 0;                                   // (B,L,128)
static constexpr long OFF_E    = OFF_V    + (long)BB*LL*128;          // (B,L,K,128)
static constexpr long OFF_EIDX = OFF_E    + (long)BB*LL*KK*128;       // (B,L,K)
static constexpr long OFF_YN   = OFF_EIDX + (long)BB*LL*KK;           // (B,L,M,128)
static constexpr long OFF_YE   = OFF_YN   + (long)BB*LL*MM*128;       // (B,L,M,M,128)
static constexpr long OFF_CTX  = OFF_YE   + (long)BB*LL*MM*MM*128;    // (B,L,M,128)

// ---------------- periodic-table lookup tables (clip at 119) --------------
__constant__ int PT_GROUP[120] = {
  0,1,18,
  1,2,13,14,15,16,17,18,
  1,2,13,14,15,16,17,18,
  1,2,3,4,5,6,7,8,9,10,11,12,13,14,15,16,17,18,
  1,2,3,4,5,6,7,8,9,10,11,12,13,14,15,16,17,18,
  1,2,3,3,3,3,3,3,3,3,3,3,3,3,3,3,3,4,5,6,7,8,9,10,11,12,13,14,15,16,17,18,
  1,2,3,3,3,3,3,3,3,3,3,3,3,3,3,3,3,4,5,6,7,8,9,10,11,12,13,14,15,16,17,18,
  18};
__constant__ int PT_PERIOD[120] = {
  0,1,1,
  2,2,2,2,2,2,2,2,
  3,3,3,3,3,3,3,3,
  4,4,4,4,4,4,4,4,4,4,4,4,4,4,4,4,4,4,
  5,5,5,5,5,5,5,5,5,5,5,5,5,5,5,5,5,5,
  6,6,6,6,6,6,6,6,6,6,6,6,6,6,6,6,6,6,6,6,6,6,6,6,6,6,6,6,6,6,6,6,
  7,7,7,7,7,7,7,7,7,7,7,7,7,7,7,7,7,7,7,7,7,7,7,7,7,7,7,7,7,7,7,7,
  7};

// ---------------- WMMA fragment helpers (ISA 7.12.2 16-bit layouts) -------
__device__ inline v16h load_a_lds(const _Float16* base, int ldk, int k0, int lane) {
  int m = lane & 15, half = lane >> 4;
  v16h a;
#pragma unroll
  for (int v = 0; v < 8; ++v) {
    int k = k0 + ((v >> 2) << 4) + (half << 3) + ((v & 3) << 1);
    a[2*v]   = base[m*ldk + k];
    a[2*v+1] = base[m*ldk + k + 1];
  }
  return a;
}

__device__ inline v16h load_b_w(const float* W, int ldn, int kdim, int k0, int n0, int lane) {
  int n = n0 + (lane & 15), half = lane >> 4;
  v16h b;
#pragma unroll
  for (int v = 0; v < 8; ++v) {
    int k = k0 + ((v >> 2) << 4) + (half << 3) + ((v & 3) << 1);
    b[2*v]   = (k     < kdim) ? (_Float16)W[(long)k*ldn + n]     : (_Float16)0.0f;
    b[2*v+1] = (k + 1 < kdim) ? (_Float16)W[(long)(k+1)*ldn + n] : (_Float16)0.0f;
  }
  return b;
}

__device__ inline void compute_a5(const float* Xr, float a5[5][3]) {
  float N[3], Ca[3], C[3], O[3], bv[3], cv[3], av[3];
  for (int i = 0; i < 3; ++i) { N[i]=Xr[i]; Ca[i]=Xr[3+i]; C[i]=Xr[6+i]; O[i]=Xr[9+i]; }
  for (int i = 0; i < 3; ++i) { bv[i]=Ca[i]-N[i]; cv[i]=C[i]-Ca[i]; }
  av[0]=bv[1]*cv[2]-bv[2]*cv[1];
  av[1]=bv[2]*cv[0]-bv[0]*cv[2];
  av[2]=bv[0]*cv[1]-bv[1]*cv[0];
  for (int i = 0; i < 3; ++i) {
    a5[0][i]=N[i]; a5[1][i]=Ca[i]; a5[2][i]=C[i]; a5[3][i]=O[i];
    a5[4][i]=-0.58273431f*av[i]+0.56802827f*bv[i]-0.54067466f*cv[i]+Ca[i];
  }
}

__device__ inline float rbf_val(float d, int q) {
  float mu = 20.0f * (float)q / 15.0f;
  float z = (d - mu) * 0.8f;           // sigma = 20/16 = 1.25
  return __expf(-z * z);
}

// ======================= 1) top-K neighbor selection ======================
__global__ void topk_kernel(const float* __restrict__ X,
                            const float* __restrict__ mask,
                            float* __restrict__ out) {
  __shared__ float dist[LL];
  __shared__ float rv[256];
  __shared__ int   ri[256];
  int bl = blockIdx.x;                      // b*L + l
  int b  = bl / LL;
  int t  = threadIdx.x;
  const float* pc = X + ((long)bl*4 + 1)*3; // Ca of center
  float cax = pc[0], cay = pc[1], caz = pc[2];
  float mask_l = mask[bl];

  float lmax = 0.0f;
  for (int j = t; j < LL; j += 256) {
    const float* p = X + (((long)b*LL + j)*4 + 1)*3;
    float dx = cax-p[0], dy = cay-p[1], dz = caz-p[2];
    float d  = sqrtf(dx*dx + dy*dy + dz*dz + 1e-6f);
    float m2 = mask_l * mask[b*LL + j];
    float df = m2 * d;
    dist[j] = df;
    lmax = fmaxf(lmax, df);
  }
  rv[t] = lmax; __syncthreads();
  for (int s = 128; s > 0; s >>= 1) { if (t < s) rv[t] = fmaxf(rv[t], rv[t+s]); __syncthreads(); }
  float rowmax = rv[0];
  __syncthreads();
  for (int j = t; j < LL; j += 256) {
    float m2 = mask_l * mask[b*LL + j];
    dist[j] += (1.0f - m2) * rowmax;
  }
  __syncthreads();

  for (int kk = 0; kk < KK; ++kk) {
    float bv = 3.4e38f; int bi = LL;
    for (int j = t; j < LL; j += 256) {
      float v = dist[j];
      if (v < bv || (v == bv && j < bi)) { bv = v; bi = j; }
    }
    rv[t] = bv; ri[t] = bi; __syncthreads();
    for (int s = 128; s > 0; s >>= 1) {
      if (t < s) {
        if (rv[t+s] < rv[t] || (rv[t+s] == rv[t] && ri[t+s] < ri[t])) {
          rv[t] = rv[t+s]; ri[t] = ri[t+s];
        }
      }
      __syncthreads();
    }
    if (t == 0) {
      int w = ri[0];
      out[OFF_EIDX + (long)bl*KK + kk] = (float)w;
      dist[w] = 3.4e38f;
    }
    __syncthreads();
  }
}

// ======================= 2) V = LN(W_node[S]) =============================
__global__ void vnode_kernel(const int* __restrict__ S,
                             const float* __restrict__ W_node,
                             const float* __restrict__ g_n,
                             const float* __restrict__ b_n,
                             float* __restrict__ out) {
  __shared__ float red[128];
  int row = blockIdx.x, t = threadIdx.x;
  float x = W_node[(long)S[row]*128 + t];
  red[t] = x; __syncthreads();
  for (int s = 64; s > 0; s >>= 1) { if (t < s) red[t] += red[t+s]; __syncthreads(); }
  float mu = red[0] * (1.0f/128.0f); __syncthreads();
  float dd = x - mu; red[t] = dd*dd; __syncthreads();
  for (int s = 64; s > 0; s >>= 1) { if (t < s) red[t] += red[t+s]; __syncthreads(); }
  float rs = rsqrtf(red[0]*(1.0f/128.0f) + 1e-5f);
  out[OFF_V + (long)row*128 + t] = dd*rs*g_n[t] + b_n[t];
}

// ======================= 3) Y_nodes = LN(3-hot @ W_yn) ====================
__global__ void ynode_kernel(const int* __restrict__ Y_t,
                             const float* __restrict__ W_yn,
                             const float* __restrict__ g_yn,
                             const float* __restrict__ b_yn,
                             float* __restrict__ out) {
  __shared__ float red[128];
  int row = blockIdx.x, t = threadIdx.x;
  int tt = Y_t[row]; if (tt > 119) tt = 119; if (tt < 0) tt = 0;
  int g = PT_GROUP[tt], p = PT_PERIOD[tt];
  float x = W_yn[(long)tt*128 + t] + W_yn[(long)(120+g)*128 + t] + W_yn[(long)(139+p)*128 + t];
  red[t] = x; __syncthreads();
  for (int s = 64; s > 0; s >>= 1) { if (t < s) red[t] += red[t+s]; __syncthreads(); }
  float mu = red[0] * (1.0f/128.0f); __syncthreads();
  float dd = x - mu; red[t] = dd*dd; __syncthreads();
  for (int s = 64; s > 0; s >>= 1) { if (t < s) red[t] += red[t+s]; __syncthreads(); }
  float rs = rsqrtf(red[0]*(1.0f/128.0f) + 1e-5f);
  out[OFF_YN + (long)row*128 + t] = dd*rs*g_yn[t] + b_yn[t];
}

// ======================= 4) E edges: WMMA 416 -> 128 + LN =================
__global__ void edge_kernel(const float* __restrict__ X,
                            const int* __restrict__ R_idx,
                            const int* __restrict__ chain,
                            const float* __restrict__ W_pos,
                            const float* __restrict__ b_pos,
                            const float* __restrict__ W_edge,
                            const float* __restrict__ g_e,
                            const float* __restrict__ b_e,
                            float* __restrict__ out) {
  __shared__ _Float16 feat[16][416];
  __shared__ float    outt[16][128];
  __shared__ int      nbr[16];
  __shared__ float    posL[16][5][3];
  __shared__ float    posN[16][5][3];
  __shared__ float    mu_s[16], rs_s[16];

  int t = threadIdx.x;
  int r = t >> 4, s = t & 15;
  long e   = (long)blockIdx.x*16 + r;
  int  b   = (int)(e / ((long)LL*KK));
  int  l   = (int)((e / KK) % LL);

  if (s == 0) nbr[r] = (int)out[OFF_EIDX + e];
  __syncthreads();
  int jn = nbr[r];

  if (s == 0) {
    float a5[5][3]; compute_a5(X + ((long)b*LL + l)*12, a5);
    for (int i = 0; i < 5; ++i) for (int q = 0; q < 3; ++q) posL[r][i][q] = a5[i][q];
  } else if (s == 1) {
    float a5[5][3]; compute_a5(X + ((long)b*LL + jn)*12, a5);
    for (int i = 0; i < 5; ++i) for (int q = 0; q < 3; ++q) posN[r][i][q] = a5[i][q];
  } else if (s == 2) {
    int off = R_idx[b*LL + l] - R_idx[b*LL + jn];
    int ec  = (chain[b*LL + l] == chain[b*LL + jn]) ? 1 : 0;
    int dcl = off + 32; if (dcl < 0) dcl = 0; if (dcl > 64) dcl = 64;
    int dpos = ec ? dcl : 65;
    for (int c = 0; c < 16; ++c)
      feat[r][c] = (_Float16)(W_pos[dpos*16 + c] + b_pos[c]);
  }
  __syncthreads();

  // 25 atom-pair distances -> 400 RBFs (each sub-thread handles up to 2 pairs)
  for (int pp = 0; pp < 2; ++pp) {
    int p = s + pp*16;
    if (p < 25) {
      int i = p / 5, j = p % 5;
      float dx = posL[r][i][0]-posN[r][j][0];
      float dy = posL[r][i][1]-posN[r][j][1];
      float dz = posL[r][i][2]-posN[r][j][2];
      float d  = sqrtf(dx*dx + dy*dy + dz*dz + 1e-6f);
      for (int q = 0; q < 16; ++q)
        feat[r][16 + p*16 + q] = (_Float16)rbf_val(d, q);
    }
  }
  __syncthreads();

  // WMMA: 16 rows x 416 K x 128 N; each wave owns a 16-col tile
  int wave = t >> 5, lane = t & 31;
  int n0 = wave * 16;
  v8f acc = {};
  for (int kt = 0; kt < 13; ++kt) {
    v16h a  = load_a_lds(&feat[0][0], 416, kt*32, lane);
    v16h bf = load_b_w(W_edge, 128, 416, kt*32, n0, lane);
    acc = __builtin_amdgcn_wmma_f32_16x16x32_f16(false, a, false, bf, (short)0, acc, false, false);
  }
  int nn = lane & 15, mb = (lane >> 4) * 8;
#pragma unroll
  for (int v = 0; v < 8; ++v) outt[mb + v][n0 + nn] = acc[v];
  __syncthreads();

  if (t < 16) {
    float sum = 0.0f;
    for (int c = 0; c < 128; ++c) sum += outt[t][c];
    float mu = sum * (1.0f/128.0f);
    float var = 0.0f;
    for (int c = 0; c < 128; ++c) { float dd = outt[t][c]-mu; var += dd*dd; }
    mu_s[t] = mu; rs_s[t] = rsqrtf(var*(1.0f/128.0f) + 1e-5f);
  }
  __syncthreads();
  {
    int c0 = s * 8;
    long base = OFF_E + e*128;
    for (int c = c0; c < c0+8; ++c)
      out[base + c] = (outt[r][c]-mu_s[r])*rs_s[r]*g_e[c] + b_e[c];
  }
}

// ======================= 5) E_context: WMMA 148 -> 128 + bias + LN ========
__global__ void ctx_kernel(const float* __restrict__ X,
                           const float* __restrict__ Y,
                           const int* __restrict__ Y_t,
                           const float* __restrict__ W_type,
                           const float* __restrict__ b_type,
                           const float* __restrict__ W_proj,
                           const float* __restrict__ b_proj,
                           const float* __restrict__ g_ctx,
                           const float* __restrict__ b_ctx,
                           float* __restrict__ out) {
  __shared__ _Float16 feat[16][160];
  __shared__ float    outt[16][128];
  __shared__ float    mu_s[16], rs_s[16];

  int bl = blockIdx.x;                 // b*L + l
  int t = threadIdx.x, m = t >> 4, s = t & 15;
  const float* Xr = X + (long)bl*12;
  const float* ym = Y + ((long)bl*16 + m)*3;
  float yx = ym[0], yy = ym[1], yz = ym[2];

  if (s < 5) {
    float ax, ay, az;
    if (s < 4) { ax = Xr[s*3]; ay = Xr[s*3+1]; az = Xr[s*3+2]; }
    else { float a5[5][3]; compute_a5(Xr, a5); ax = a5[4][0]; ay = a5[4][1]; az = a5[4][2]; }
    float dx = ax-yx, dy = ay-yy, dz = az-yz;
    float d  = sqrtf(dx*dx + dy*dy + dz*dz + 1e-6f);
    for (int q = 0; q < 16; ++q) feat[m][s*16 + q] = (_Float16)rbf_val(d, q);
  } else if (s == 5) {
    // local frame + angle features
    float cax = Xr[3], cay = Xr[4], caz = Xr[5];
    float v1x = Xr[0]-cax, v1y = Xr[1]-cay, v1z = Xr[2]-caz;   // N - Ca
    float v2x = Xr[6]-cax, v2y = Xr[7]-cay, v2z = Xr[8]-caz;   // C - Ca
    float n1 = fmaxf(sqrtf(v1x*v1x+v1y*v1y+v1z*v1z), 1e-12f);
    float e1x = v1x/n1, e1y = v1y/n1, e1z = v1z/n1;
    float dp  = e1x*v2x + e1y*v2y + e1z*v2z;
    float u2x = v2x-e1x*dp, u2y = v2y-e1y*dp, u2z = v2z-e1z*dp;
    float n2 = fmaxf(sqrtf(u2x*u2x+u2y*u2y+u2z*u2z), 1e-12f);
    float e2x = u2x/n2, e2y = u2y/n2, e2z = u2z/n2;
    float e3x = e1y*e2z - e1z*e2y;
    float e3y = e1z*e2x - e1x*e2z;
    float e3z = e1x*e2y - e1y*e2x;
    float dx = yx-cax, dy = yy-cay, dz = yz-caz;
    float l0 = e1x*dx + e1y*dy + e1z*dz;
    float l1 = e2x*dx + e2y*dy + e2z*dz;
    float l2 = e3x*dx + e3y*dy + e3z*dz;
    float rxy  = sqrtf(l0*l0 + l1*l1 + 1e-8f);
    float rxyz = sqrtf(l0*l0 + l1*l1 + l2*l2) + 1e-8f;
    feat[m][144] = (_Float16)(l0/rxy);
    feat[m][145] = (_Float16)(l1/rxy);
    feat[m][146] = (_Float16)(rxy/rxyz);
    feat[m][147] = (_Float16)(l2/rxyz);
  } else if (s == 6) {
    for (int c = 148; c < 160; ++c) feat[m][c] = (_Float16)0.0f;
  }
  // 64-dim type embedding, 4 values per sub-thread
  {
    int tt = Y_t[(long)bl*16 + m]; if (tt > 119) tt = 119; if (tt < 0) tt = 0;
    int g = PT_GROUP[tt], p = PT_PERIOD[tt];
    for (int q = 0; q < 4; ++q) {
      int c = s*4 + q;
      feat[m][80 + c] = (_Float16)(W_type[(long)tt*64 + c] + W_type[(long)(120+g)*64 + c]
                                 + W_type[(long)(139+p)*64 + c] + b_type[c]);
    }
  }
  __syncthreads();

  int wave = t >> 5, lane = t & 31;
  int n0 = wave * 16;
  v8f acc = {};
  for (int kt = 0; kt < 5; ++kt) {
    v16h a  = load_a_lds(&feat[0][0], 160, kt*32, lane);
    v16h bf = load_b_w(W_proj, 128, 148, kt*32, n0, lane);
    acc = __builtin_amdgcn_wmma_f32_16x16x32_f16(false, a, false, bf, (short)0, acc, false, false);
  }
  int nn = lane & 15, mb = (lane >> 4) * 8;
#pragma unroll
  for (int v = 0; v < 8; ++v) outt[mb + v][n0 + nn] = acc[v] + b_proj[n0 + nn];
  __syncthreads();

  if (t < 16) {
    float sum = 0.0f;
    for (int c = 0; c < 128; ++c) sum += outt[t][c];
    float mu = sum * (1.0f/128.0f);
    float var = 0.0f;
    for (int c = 0; c < 128; ++c) { float dd = outt[t][c]-mu; var += dd*dd; }
    mu_s[t] = mu; rs_s[t] = rsqrtf(var*(1.0f/128.0f) + 1e-5f);
  }
  __syncthreads();
  {
    int c0 = s * 8;
    long base = OFF_CTX + ((long)bl*16 + m)*128;
    for (int c = c0; c < c0+8; ++c)
      out[base + c] = (outt[m][c]-mu_s[m])*rs_s[m]*g_ctx[c] + b_ctx[c];
  }
}

// ======================= 6) Y_edges: WMMA 16 -> 128 + LN (big output) =====
__global__ void yedge_kernel(const float* __restrict__ Y,
                             const float* __restrict__ W_ye,
                             const float* __restrict__ g_ye,
                             const float* __restrict__ b_ye,
                             float* __restrict__ out) {
  __shared__ _Float16 feat[16][32];
  __shared__ float    outt[16][128];
  __shared__ float    mu_s[16], rs_s[16];

  int bid = blockIdx.x;                // (b*L + l)*16 + mi
  int bl = bid >> 4, mi = bid & 15;
  int t = threadIdx.x, m = t >> 4, s = t & 15;   // m = mj
  const float* yi = Y + ((long)bl*16 + mi)*3;
  const float* yj = Y + ((long)bl*16 + m)*3;
  float dx = yi[0]-yj[0], dy = yi[1]-yj[1], dz = yi[2]-yj[2];
  float d  = sqrtf(dx*dx + dy*dy + dz*dz + 1e-6f);
  feat[m][s]      = (_Float16)rbf_val(d, s);
  feat[m][16 + s] = (_Float16)0.0f;
  __syncthreads();

  int wave = t >> 5, lane = t & 31;
  int n0 = wave * 16;
  v8f acc = {};
  {
    v16h a  = load_a_lds(&feat[0][0], 32, 0, lane);
    v16h bf = load_b_w(W_ye, 128, 16, 0, n0, lane);
    acc = __builtin_amdgcn_wmma_f32_16x16x32_f16(false, a, false, bf, (short)0, acc, false, false);
  }
  int nn = lane & 15, mb = (lane >> 4) * 8;
#pragma unroll
  for (int v = 0; v < 8; ++v) outt[mb + v][n0 + nn] = acc[v];
  __syncthreads();

  if (t < 16) {
    float sum = 0.0f;
    for (int c = 0; c < 128; ++c) sum += outt[t][c];
    float mu = sum * (1.0f/128.0f);
    float var = 0.0f;
    for (int c = 0; c < 128; ++c) { float dd = outt[t][c]-mu; var += dd*dd; }
    mu_s[t] = mu; rs_s[t] = rsqrtf(var*(1.0f/128.0f) + 1e-5f);
  }
  __syncthreads();
  {
    int c0 = s * 8;
    long base = OFF_YE + ((long)bid*16 + m)*128;
    for (int c = c0; c < c0+8; ++c)
      out[base + c] = (outt[m][c]-mu_s[m])*rs_s[m]*g_ye[c] + b_ye[c];
  }
}

// =========================================================================
extern "C" void kernel_launch(void* const* d_in, const int* in_sizes, int n_in,
                              void* d_out, int out_size, void* d_ws, size_t ws_size,
                              hipStream_t stream) {
  (void)in_sizes; (void)n_in; (void)out_size; (void)d_ws; (void)ws_size;
  const float* X      = (const float*)d_in[0];
  const float* Y      = (const float*)d_in[1];
  const float* mask   = (const float*)d_in[3];
  const int*   S      = (const int*)  d_in[4];
  const int*   Y_t    = (const int*)  d_in[5];
  const int*   R_idx  = (const int*)  d_in[6];
  const int*   chain  = (const int*)  d_in[7];
  const float* W_pos  = (const float*)d_in[8];
  const float* b_pos  = (const float*)d_in[9];
  const float* W_edge = (const float*)d_in[10];
  const float* g_e    = (const float*)d_in[11];
  const float* b_e    = (const float*)d_in[12];
  const float* W_node = (const float*)d_in[13];
  const float* g_n    = (const float*)d_in[14];
  const float* b_n    = (const float*)d_in[15];
  const float* W_type = (const float*)d_in[16];
  const float* b_type = (const float*)d_in[17];
  const float* W_proj = (const float*)d_in[18];
  const float* b_proj = (const float*)d_in[19];
  const float* g_ctx  = (const float*)d_in[20];
  const float* b_ctx  = (const float*)d_in[21];
  const float* W_ye   = (const float*)d_in[22];
  const float* g_ye   = (const float*)d_in[23];
  const float* b_ye   = (const float*)d_in[24];
  const float* W_yn   = (const float*)d_in[25];
  const float* g_yn   = (const float*)d_in[26];
  const float* b_yn   = (const float*)d_in[27];
  float* out = (float*)d_out;

  topk_kernel <<<BB*LL,            256, 0, stream>>>(X, mask, out);
  vnode_kernel<<<BB*LL,            128, 0, stream>>>(S, W_node, g_n, b_n, out);
  ynode_kernel<<<BB*LL*MM,         128, 0, stream>>>(Y_t, W_yn, g_yn, b_yn, out);
  edge_kernel <<<(BB*LL*KK)/16,    256, 0, stream>>>(X, R_idx, chain, W_pos, b_pos,
                                                     W_edge, g_e, b_e, out);
  ctx_kernel  <<<BB*LL,            256, 0, stream>>>(X, Y, Y_t, W_type, b_type,
                                                     W_proj, b_proj, g_ctx, b_ctx, out);
  yedge_kernel<<<BB*LL*MM,         256, 0, stream>>>(Y, W_ye, g_ye, b_ye, out);
}
